// Model_36412732735939
// MI455X (gfx1250) — compile-verified, matrix operand order checked by reading, not run on hardware
//
#include <hip/hip_runtime.h>
#include <hip/hip_bf16.h>

// ---------------------------------------------------------------------------
// Model dims
// ---------------------------------------------------------------------------
#define DD   768
#define HH   384
#define BB   8
#define KK   24
#define LL   12
#define PP   20
#define VV   44

#define N1   (BB*KK*LL)   // 2304 sequences for stage-1 GRU
#define N2   (BB*KK)      // 192 sequences for stage-3 BiGRU
#define G1   (3*DD)       // 2304 gate cols for GRU(768)
#define G2   (3*HH)       // 1152 gate cols for GRU(384)
#define G4   (4*HH)       // 1536 gate cols for LSTM(384)

// ---------------------------------------------------------------------------
// WMMA types / helpers (gfx1250, wave32)
// ---------------------------------------------------------------------------
typedef __attribute__((ext_vector_type(16))) __bf16 v16bf;
typedef __attribute__((ext_vector_type(8)))  float  v8f;

struct BFrag { uint4 lo, hi; };

__device__ __forceinline__ unsigned short f2bf_rte(float x) {
    unsigned int u = __builtin_bit_cast(unsigned int, x);
    unsigned int r = (u + 0x7FFFu + ((u >> 16) & 1u)) >> 16;
    return (unsigned short)r;
}
__device__ __forceinline__ float bf2f(unsigned short b) {
    unsigned int u = ((unsigned int)b) << 16;
    return __builtin_bit_cast(float, u);
}
__device__ __forceinline__ float sigf(float x) {
    return 1.0f / (1.0f + __expf(-x));
}

// Load one 16x32 bf16 fragment (A-style layout) for this lane from a
// row-major [rows, ld] bf16 matrix. lane<16 holds K kb+0..7 / kb+16..23,
// lane>=16 holds K kb+8..15 / kb+24..31, row = (lane&15)-th row of the tile.
__device__ __forceinline__ v16bf load_frag(const unsigned short* __restrict__ base,
                                           long row, int ld, int kb, int half) {
    const unsigned short* p = base + row * (long)ld + kb + half * 8;
    BFrag f;
    f.lo = *(const uint4*)(p);
    f.hi = *(const uint4*)(p + 16);
    return __builtin_bit_cast(v16bf, f);
}

// ---------------------------------------------------------------------------
// Elementwise utility kernels
// ---------------------------------------------------------------------------
__global__ void f2bf_kernel(const float* __restrict__ in, unsigned short* __restrict__ out, long n) {
    long i = (long)blockIdx.x * blockDim.x + threadIdx.x;
    if (i < n) out[i] = f2bf_rte(in[i]);
}
__global__ void fill_f32_kernel(float* __restrict__ p, float v, long n) {
    long i = (long)blockIdx.x * blockDim.x + threadIdx.x;
    if (i < n) p[i] = v;
}
__global__ void fill_u16_kernel(unsigned short* __restrict__ p, unsigned short v, long n) {
    long i = (long)blockIdx.x * blockDim.x + threadIdx.x;
    if (i < n) p[i] = v;
}

// maxbuf [192,768] (row = b*KK+k) -> relu -> subT bf16 [192,768] (row = k*BB+b)
__global__ void maxrelu_reorder_kernel(const float* __restrict__ mb, unsigned short* __restrict__ subT) {
    int idx = blockIdx.x * blockDim.x + threadIdx.x;
    if (idx >= N2 * (2 * HH)) return;
    int r = idx / (2 * HH), j = idx % (2 * HH);
    int b = r / KK, k = r % KK;
    float v = mb[idx];
    v = v > 0.f ? v : 0.f;
    subT[(long)(k * BB + b) * (2 * HH) + j] = f2bf_rte(v);
}

// ---------------------------------------------------------------------------
// Generic WMMA GEMM: C[m,n] = sum_k A[m,k]*B[n,k] (+bias[n]) (+add[m,n])
// A: [M,K] bf16 row-major, B: [N,K] bf16 row-major. One wave -> 16x64 tile.
// N must be a multiple of 64, K a multiple of 32.
// ---------------------------------------------------------------------------
__global__ void wmma_gemm_kernel(const unsigned short* __restrict__ A, int lda,
                                 const unsigned short* __restrict__ Bw, int ldb,
                                 const float* __restrict__ bias,
                                 const float* __restrict__ addsrc,
                                 float* __restrict__ Cf,
                                 unsigned short* __restrict__ Cb,
                                 int M, int N, int K) {
    const int lane = threadIdx.x & 31;
    const int wave = threadIdx.x >> 5;
    const int nT = N >> 6;
    const int mT = (M + 15) >> 4;
    const int tile = blockIdx.x * (blockDim.x >> 5) + wave;
    if (tile >= nT * mT) return;
    const int mt = tile / nT, nt = tile % nT;
    const int half = lane >> 4, l16 = lane & 15;

    long arow = (long)mt * 16 + l16;
    if (arow >= M) arow = M - 1;   // clamp (padded rows never stored)

    v8f acc[4] = {{}, {}, {}, {}};
    for (int kb = 0; kb < K; kb += 32) {
        v16bf a = load_frag(A, arow, lda, kb, half);
#pragma unroll
        for (int q = 0; q < 4; ++q) {
            v16bf b = load_frag(Bw, (long)(nt * 64 + q * 16 + l16), ldb, kb, half);
            acc[q] = __builtin_amdgcn_wmma_f32_16x16x32_bf16(
                false, a, false, b, (short)0, acc[q], false, false);
        }
    }
#pragma unroll
    for (int q = 0; q < 4; ++q) {
        int col = nt * 64 + q * 16 + l16;
        float bv = bias ? bias[col] : 0.f;
#pragma unroll
        for (int v = 0; v < 8; ++v) {
            int row = mt * 16 + half * 8 + v;
            if (row >= M) continue;
            float x = acc[q][v] + bv;
            long o = (long)row * N + col;
            if (addsrc) x += addsrc[o];
            if (Cf) Cf[o] = x;
            if (Cb) Cb[o] = f2bf_rte(x);
        }
    }
}

// ---------------------------------------------------------------------------
// Fused GRU step: h' = (1-z)*tanh(gi_n + r*(h@Whh_n^T + bhh_n)) + z*h
// r = sig(gi_r + gh_r + bhh_r), z = sig(gi_z + gh_z + bhh_z)
// gi from LUT-gather (stage 1) or strided precomputed buffer (stage 3).
// One wave -> 32 rows x 16 hidden cols: 2 A-frags reuse the 3 gate B-frags
// (6 WMMA per K-iter, halves Whh traffic in the dominant stage-1 GEMM).
// ---------------------------------------------------------------------------
__global__ void gru_step_kernel(const unsigned short* __restrict__ hin,
                                unsigned short* __restrict__ hout,
                                const unsigned short* __restrict__ Whh,
                                const float* __restrict__ bhh,
                                const float* __restrict__ gi, long giStride, long giOff,
                                const float* __restrict__ lut,
                                const int* __restrict__ toks, int tokStride, int tokOff,
                                float* __restrict__ hmax, int hmaxLd, int hmaxOff,
                                float* __restrict__ hsout,
                                int Nb, int Hd) {
    const int lane = threadIdx.x & 31;
    const int wave = threadIdx.x >> 5;
    const int jT = Hd >> 4;
    const int mT = (Nb + 31) >> 5;     // 32-row tiles
    const int tile = blockIdx.x * (blockDim.x >> 5) + wave;
    if (tile >= jT * mT) return;
    const int mt = tile / jT, jt = tile % jT;
    const int half = lane >> 4, l16 = lane & 15;

    long arow0 = (long)mt * 32 + l16;
    long arow1 = arow0 + 16;
    if (arow0 >= Nb) arow0 = Nb - 1;
    if (arow1 >= Nb) arow1 = Nb - 1;
    const long brow = (long)jt * 16 + l16;

    v8f accr[2] = {{}, {}}, accz[2] = {{}, {}}, accn[2] = {{}, {}};
    for (int kb = 0; kb < Hd; kb += 32) {
        v16bf a0 = load_frag(hin, arow0, Hd, kb, half);
        v16bf a1 = load_frag(hin, arow1, Hd, kb, half);
        v16bf br = load_frag(Whh, brow,            Hd, kb, half);
        v16bf bz = load_frag(Whh, (long)Hd + brow, Hd, kb, half);
        v16bf bn = load_frag(Whh, 2L * Hd + brow,  Hd, kb, half);
        accr[0] = __builtin_amdgcn_wmma_f32_16x16x32_bf16(false, a0, false, br, (short)0, accr[0], false, false);
        accz[0] = __builtin_amdgcn_wmma_f32_16x16x32_bf16(false, a0, false, bz, (short)0, accz[0], false, false);
        accn[0] = __builtin_amdgcn_wmma_f32_16x16x32_bf16(false, a0, false, bn, (short)0, accn[0], false, false);
        accr[1] = __builtin_amdgcn_wmma_f32_16x16x32_bf16(false, a1, false, br, (short)0, accr[1], false, false);
        accz[1] = __builtin_amdgcn_wmma_f32_16x16x32_bf16(false, a1, false, bz, (short)0, accz[1], false, false);
        accn[1] = __builtin_amdgcn_wmma_f32_16x16x32_bf16(false, a1, false, bn, (short)0, accn[1], false, false);
    }

    const int j = jt * 16 + l16;
    const float bhr = bhh[j], bhz = bhh[Hd + j], bhn = bhh[2 * Hd + j];
#pragma unroll
    for (int rh = 0; rh < 2; ++rh) {
#pragma unroll
        for (int v = 0; v < 8; ++v) {
            int row = mt * 32 + rh * 16 + half * 8 + v;
            if (row >= Nb) continue;
            float gr, gz, gn;
            if (lut) {
                int tok = toks[(long)row * tokStride + tokOff];
                const float* lp = lut + (long)tok * (3 * Hd);
                gr = lp[j]; gz = lp[Hd + j]; gn = lp[2 * Hd + j];
            } else {
                const float* gp = gi + (long)row * giStride + giOff;
                gr = gp[j]; gz = gp[Hd + j]; gn = gp[2 * Hd + j];
            }
            float hprev = bf2f(hin[(long)row * Hd + j]);
            float r = sigf(gr + accr[rh][v] + bhr);
            float z = sigf(gz + accz[rh][v] + bhz);
            float n = tanhf(gn + r * (accn[rh][v] + bhn));
            float hnew = (1.f - z) * n + z * hprev;
            hout[(long)row * Hd + j] = f2bf_rte(hnew);
            if (hmax) {
                float* mp = hmax + (long)row * hmaxLd + hmaxOff + j;
                *mp = fmaxf(*mp, hnew);
            }
            if (hsout) hsout[(long)row * Hd + j] = hnew;
        }
    }
}

// ---------------------------------------------------------------------------
// Fused LSTM step (gates i,f,g,o): c' = sig(f)*c + sig(i)*tanh(g);
// h' = sig(o)*tanh(c'). Running max into hmax. 32 rows x 16 cols per wave.
// ---------------------------------------------------------------------------
__global__ void lstm_step_kernel(const unsigned short* __restrict__ hin,
                                 unsigned short* __restrict__ hout,
                                 const float* __restrict__ cin,
                                 float* __restrict__ cout,
                                 const unsigned short* __restrict__ Whh,
                                 const float* __restrict__ bhh,
                                 const float* __restrict__ gi, long giStride, long giOff,
                                 float* __restrict__ hmax, int hmaxLd, int hmaxOff,
                                 int Nb, int Hd) {
    const int lane = threadIdx.x & 31;
    const int wave = threadIdx.x >> 5;
    const int jT = Hd >> 4;
    const int mT = (Nb + 31) >> 5;
    const int tile = blockIdx.x * (blockDim.x >> 5) + wave;
    if (tile >= jT * mT) return;
    const int mt = tile / jT, jt = tile % jT;
    const int half = lane >> 4, l16 = lane & 15;

    long arow0 = (long)mt * 32 + l16;
    long arow1 = arow0 + 16;
    if (arow0 >= Nb) arow0 = Nb - 1;
    if (arow1 >= Nb) arow1 = Nb - 1;
    const long brow = (long)jt * 16 + l16;

    v8f acc[2][4] = {{{}, {}, {}, {}}, {{}, {}, {}, {}}};
    for (int kb = 0; kb < Hd; kb += 32) {
        v16bf a0 = load_frag(hin, arow0, Hd, kb, half);
        v16bf a1 = load_frag(hin, arow1, Hd, kb, half);
#pragma unroll
        for (int g = 0; g < 4; ++g) {
            v16bf b = load_frag(Whh, (long)g * Hd + brow, Hd, kb, half);
            acc[0][g] = __builtin_amdgcn_wmma_f32_16x16x32_bf16(false, a0, false, b, (short)0, acc[0][g], false, false);
            acc[1][g] = __builtin_amdgcn_wmma_f32_16x16x32_bf16(false, a1, false, b, (short)0, acc[1][g], false, false);
        }
    }

    const int j = jt * 16 + l16;
    const float bi = bhh[j], bf = bhh[Hd + j], bg = bhh[2 * Hd + j], bo = bhh[3 * Hd + j];
#pragma unroll
    for (int rh = 0; rh < 2; ++rh) {
#pragma unroll
        for (int v = 0; v < 8; ++v) {
            int row = mt * 32 + rh * 16 + half * 8 + v;
            if (row >= Nb) continue;
            const float* gp = gi + (long)row * giStride + giOff;
            float i_ = sigf(gp[j]           + acc[rh][0][v] + bi);
            float f_ = sigf(gp[Hd + j]      + acc[rh][1][v] + bf);
            float g_ = tanhf(gp[2 * Hd + j] + acc[rh][2][v] + bg);
            float o_ = sigf(gp[3 * Hd + j]  + acc[rh][3][v] + bo);
            float cprev = cin[(long)row * Hd + j];
            float cnew = f_ * cprev + i_ * g_;
            float hnew = o_ * tanhf(cnew);
            cout[(long)row * Hd + j] = cnew;
            hout[(long)row * Hd + j] = f2bf_rte(hnew);
            float* mp = hmax + (long)row * hmaxLd + hmaxOff + j;
            *mp = fmaxf(*mp, hnew);
        }
    }
}

// ---------------------------------------------------------------------------
// MLP head on ast_out [8,768]: BN(train) -> dense(1536)+relu -> BN -> dense(768)
// +relu -> out(1) -> sigmoid. Single block; tiny work.
// ---------------------------------------------------------------------------
__global__ void head_kernel(const float* __restrict__ ast,
                            const float* __restrict__ bn1g, const float* __restrict__ bn1b,
                            const float* __restrict__ dW, const float* __restrict__ db,
                            const float* __restrict__ bn2g, const float* __restrict__ bn2b,
                            const float* __restrict__ d2W, const float* __restrict__ d2b,
                            const float* __restrict__ oW, const float* __restrict__ ob,
                            float* __restrict__ tmp1, float* __restrict__ tmp2,
                            float* __restrict__ out) {
    const int tid = threadIdx.x;
    // BN1 over batch of 8
    for (int j = tid; j < DD; j += 256) {
        float mu = 0.f, m2 = 0.f;
        for (int i = 0; i < BB; ++i) { float x = ast[i * DD + j]; mu += x; m2 += x * x; }
        mu *= (1.f / BB);
        float var = m2 * (1.f / BB) - mu * mu;
        float inv = rsqrtf(var + 1e-5f) * bn1g[j];
        float bb = bn1b[j];
        for (int i = 0; i < BB; ++i) tmp1[i * DD + j] = (ast[i * DD + j] - mu) * inv + bb;
    }
    __syncthreads();
    // dense1 + relu -> tmp2 [8,1536]
    for (int idx = tid; idx < BB * (2 * DD); idx += 256) {
        int i = idx / (2 * DD), o = idx % (2 * DD);
        float s = db[o];
        const float* w = dW + (long)o * DD;
        const float* x = tmp1 + (long)i * DD;
        for (int j = 0; j < DD; ++j) s += x[j] * w[j];
        tmp2[idx] = s > 0.f ? s : 0.f;
    }
    __syncthreads();
    // BN2 in place on tmp2
    for (int j = tid; j < 2 * DD; j += 256) {
        float mu = 0.f, m2 = 0.f;
        for (int i = 0; i < BB; ++i) { float x = tmp2[i * (2 * DD) + j]; mu += x; m2 += x * x; }
        mu *= (1.f / BB);
        float var = m2 * (1.f / BB) - mu * mu;
        float inv = rsqrtf(var + 1e-5f) * bn2g[j];
        float bb = bn2b[j];
        for (int i = 0; i < BB; ++i)
            tmp2[i * (2 * DD) + j] = (tmp2[i * (2 * DD) + j] - mu) * inv + bb;
    }
    __syncthreads();
    // dense2 + relu -> tmp1 [8,768]
    for (int idx = tid; idx < BB * DD; idx += 256) {
        int i = idx / DD, o = idx % DD;
        float s = d2b[o];
        const float* w = d2W + (long)o * (2 * DD);
        const float* x = tmp2 + (long)i * (2 * DD);
        for (int j = 0; j < 2 * DD; ++j) s += x[j] * w[j];
        tmp1[idx] = s > 0.f ? s : 0.f;
    }
    __syncthreads();
    if (tid < BB) {
        float s = ob[0];
        const float* x = tmp1 + (long)tid * DD;
        for (int j = 0; j < DD; ++j) s += x[j] * oW[j];
        out[tid] = 1.f / (1.f + __expf(-s));
    }
}

// ---------------------------------------------------------------------------
// Host side
// ---------------------------------------------------------------------------
extern "C" void kernel_launch(void* const* d_in, const int* in_sizes, int n_in,
                              void* d_out, int out_size, void* d_ws, size_t ws_size,
                              hipStream_t stream) {
    (void)in_sizes; (void)n_in; (void)out_size; (void)ws_size;

    const float* feature_s = (const float*)d_in[0];
    const float* emb_W    = (const float*)d_in[1];
    const float* tree_W   = (const float*)d_in[2];
    const float* tree_b   = (const float*)d_in[3];
    const float* gru_Wih  = (const float*)d_in[4];
    const float* gru_Whh  = (const float*)d_in[5];
    const float* gru_bih  = (const float*)d_in[6];
    const float* gru_bhh  = (const float*)d_in[7];
    const float* g1f_Wih  = (const float*)d_in[8];
    const float* g1f_Whh  = (const float*)d_in[9];
    const float* g1f_bih  = (const float*)d_in[10];
    const float* g1f_bhh  = (const float*)d_in[11];
    const float* g1b_Wih  = (const float*)d_in[12];
    const float* g1b_Whh  = (const float*)d_in[13];
    const float* g1b_bih  = (const float*)d_in[14];
    const float* g1b_bhh  = (const float*)d_in[15];
    const float* lf_Wih   = (const float*)d_in[16];
    const float* lf_Whh   = (const float*)d_in[17];
    const float* lf_bih   = (const float*)d_in[18];
    const float* lf_bhh   = (const float*)d_in[19];
    const float* lb_Wih   = (const float*)d_in[20];
    const float* lb_Whh   = (const float*)d_in[21];
    const float* lb_bih   = (const float*)d_in[22];
    const float* lb_bhh   = (const float*)d_in[23];
    const float* bn1_g    = (const float*)d_in[24];
    const float* bn1_b    = (const float*)d_in[25];
    const float* bn2_g    = (const float*)d_in[26];
    const float* bn2_b    = (const float*)d_in[27];
    const float* dense_W  = (const float*)d_in[28];
    const float* dense_b  = (const float*)d_in[29];
    const float* dense2_W = (const float*)d_in[30];
    const float* dense2_b = (const float*)d_in[31];
    const float* out_W    = (const float*)d_in[32];
    const float* out_b    = (const float*)d_in[33];
    const int*   paths    = (const int*)d_in[34];
    float* out = (float*)d_out;

    // Bump allocator over workspace (256B aligned)
    char* wsp = (char*)d_ws;
    size_t off = 0;
    auto alloc_bytes = [&](size_t b) -> void* {
        void* p = wsp + off;
        off = (off + b + 255) & ~(size_t)255;
        return p;
    };
    auto alloc_f = [&](size_t n) -> float* { return (float*)alloc_bytes(n * 4); };
    auto alloc_h = [&](size_t n) -> unsigned short* { return (unsigned short*)alloc_bytes(n * 2); };

    // bf16 weight/activation copies
    unsigned short* emb_bf    = alloc_h((size_t)VV * DD);
    unsigned short* gruWih_bf = alloc_h((size_t)G1 * DD);
    unsigned short* gruWhh_bf = alloc_h((size_t)G1 * DD);
    unsigned short* treeW_bf  = alloc_h((size_t)DD * DD);
    unsigned short* fs_bf     = alloc_h((size_t)N1 * DD);
    unsigned short* g1fWih_bf = alloc_h((size_t)G2 * DD);
    unsigned short* g1fWhh_bf = alloc_h((size_t)G2 * HH);
    unsigned short* g1bWih_bf = alloc_h((size_t)G2 * DD);
    unsigned short* g1bWhh_bf = alloc_h((size_t)G2 * HH);
    unsigned short* lfWih_bf  = alloc_h((size_t)G4 * DD);
    unsigned short* lfWhh_bf  = alloc_h((size_t)G4 * HH);
    unsigned short* lbWih_bf  = alloc_h((size_t)G4 * DD);
    unsigned short* lbWhh_bf  = alloc_h((size_t)G4 * HH);
    unsigned short* h1a       = alloc_h((size_t)N1 * DD);
    unsigned short* h1b       = alloc_h((size_t)N1 * DD);
    unsigned short* leafs_bf  = alloc_h((size_t)N1 * DD);
    unsigned short* h3a       = alloc_h((size_t)N2 * HH);
    unsigned short* h3b       = alloc_h((size_t)N2 * HH);
    unsigned short* subT_bf   = alloc_h((size_t)N2 * (2 * HH));
    unsigned short* h4a       = alloc_h((size_t)BB * HH);
    unsigned short* h4b       = alloc_h((size_t)BB * HH);
    // fp32 buffers
    float* lut     = alloc_f((size_t)VV * G1);
    float* pathvec = alloc_f((size_t)N1 * DD);
    float* gi2f    = alloc_f((size_t)N1 * G2);
    float* gi2b    = alloc_f((size_t)N1 * G2);
    float* maxbuf  = alloc_f((size_t)N2 * (2 * HH));
    float* gi4f    = alloc_f((size_t)(KK * BB) * G4);
    float* gi4b    = alloc_f((size_t)(KK * BB) * G4);
    float* c4      = alloc_f((size_t)BB * HH);
    float* astmax  = alloc_f((size_t)BB * (2 * HH));
    float* tmp1    = alloc_f((size_t)BB * DD);
    float* tmp2    = alloc_f((size_t)BB * (2 * DD));

    auto conv = [&](const float* src, unsigned short* dst, long n) {
        int blocks = (int)((n + 255) / 256);
        f2bf_kernel<<<blocks, 256, 0, stream>>>(src, dst, n);
    };
    auto fillf = [&](float* p, float v, long n) {
        int blocks = (int)((n + 255) / 256);
        fill_f32_kernel<<<blocks, 256, 0, stream>>>(p, v, n);
    };
    auto fillh = [&](unsigned short* p, unsigned short v, long n) {
        int blocks = (int)((n + 255) / 256);
        fill_u16_kernel<<<blocks, 256, 0, stream>>>(p, v, n);
    };
    auto gemm = [&](const unsigned short* A, int lda, const unsigned short* Bw, int ldb,
                    const float* bias, const float* addsrc, float* Cf, unsigned short* Cb,
                    int M, int N, int Kd) {
        int waves = ((M + 15) / 16) * (N / 64);
        int blocks = (waves + 3) / 4;
        wmma_gemm_kernel<<<blocks, 128, 0, stream>>>(A, lda, Bw, ldb, bias, addsrc, Cf, Cb, M, N, Kd);
    };
    auto gru_step = [&](const unsigned short* hin, unsigned short* hout,
                        const unsigned short* Whh, const float* bhh,
                        const float* gi, long giStride, long giOff,
                        const float* lutp, const int* toks, int tokStride, int tokOff,
                        float* hmax, int hmaxLd, int hmaxOff, float* hsout,
                        int Nb, int Hd) {
        int waves = ((Nb + 31) / 32) * (Hd / 16);
        int blocks = (waves + 3) / 4;
        gru_step_kernel<<<blocks, 128, 0, stream>>>(hin, hout, Whh, bhh,
                                                    gi, giStride, giOff,
                                                    lutp, toks, tokStride, tokOff,
                                                    hmax, hmaxLd, hmaxOff, hsout, Nb, Hd);
    };
    auto lstm_step = [&](const unsigned short* hin, unsigned short* hout,
                         const float* cin, float* cout,
                         const unsigned short* Whh, const float* bhh,
                         const float* gi, long giStride, long giOff,
                         float* hmax, int hmaxLd, int hmaxOff, int Nb, int Hd) {
        int waves = ((Nb + 31) / 32) * (Hd / 16);
        int blocks = (waves + 3) / 4;
        lstm_step_kernel<<<blocks, 128, 0, stream>>>(hin, hout, cin, cout, Whh, bhh,
                                                     gi, giStride, giOff,
                                                     hmax, hmaxLd, hmaxOff, Nb, Hd);
    };

    // ---- 0) bf16 conversions ------------------------------------------------
    conv(emb_W,    emb_bf,    (long)VV * DD);
    conv(gru_Wih,  gruWih_bf, (long)G1 * DD);
    conv(gru_Whh,  gruWhh_bf, (long)G1 * DD);
    conv(tree_W,   treeW_bf,  (long)DD * DD);
    conv(feature_s, fs_bf,    (long)N1 * DD);
    conv(g1f_Wih,  g1fWih_bf, (long)G2 * DD);
    conv(g1f_Whh,  g1fWhh_bf, (long)G2 * HH);
    conv(g1b_Wih,  g1bWih_bf, (long)G2 * DD);
    conv(g1b_Whh,  g1bWhh_bf, (long)G2 * HH);
    conv(lf_Wih,   lfWih_bf,  (long)G4 * DD);
    conv(lf_Whh,   lfWhh_bf,  (long)G4 * HH);
    conv(lb_Wih,   lbWih_bf,  (long)G4 * DD);
    conv(lb_Whh,   lbWhh_bf,  (long)G4 * HH);

    // ---- 1) token gate LUT: emb_W @ gru_Wih^T + gru_bih -> [44, 2304] -------
    gemm(emb_bf, DD, gruWih_bf, DD, gru_bih, nullptr, lut, nullptr, VV, G1, DD);

    // ---- 2) stage-1 GRU(768) over P=20, batch 2304; gi gathered from LUT ----
    fillh(h1a, 0, (long)N1 * DD);
    for (int t = 0; t < PP; ++t) {
        float* hs = (t == PP - 1) ? pathvec : nullptr;   // final hidden
        gru_step(h1a, h1b, gruWhh_bf, gru_bhh,
                 nullptr, 0, 0, lut, paths, PP, t,
                 nullptr, 0, 0, hs, N1, DD);
        unsigned short* tmp = h1a; h1a = h1b; h1b = tmp;
    }

    // ---- 3) leafs = fs @ tree_W^T + tree_b + path_vec -> bf16 ----------------
    gemm(fs_bf, DD, treeW_bf, DD, tree_b, pathvec, nullptr, leafs_bf, N1, DD, DD);

    // ---- 4) stage-3 BiGRU(384) input gates precompute ------------------------
    gemm(leafs_bf, DD, g1fWih_bf, DD, g1f_bih, nullptr, gi2f, nullptr, N1, G2, DD);
    gemm(leafs_bf, DD, g1bWih_bf, DD, g1b_bih, nullptr, gi2b, nullptr, N1, G2, DD);

    // ---- 5) BiGRU over L=12; feature-wise max over sequence ------------------
    fillf(maxbuf, -1e30f, (long)N2 * (2 * HH));
    fillh(h3a, 0, (long)N2 * HH);
    for (int l = 0; l < LL; ++l) {   // forward direction
        gru_step(h3a, h3b, g1fWhh_bf, g1f_bhh,
                 gi2f, (long)LL * G2, (long)l * G2,
                 nullptr, nullptr, 0, 0,
                 maxbuf, 2 * HH, 0, nullptr, N2, HH);
        unsigned short* tmp = h3a; h3a = h3b; h3b = tmp;
    }
    fillh(h3a, 0, (long)N2 * HH);
    for (int s = 0; s < LL; ++s) {   // backward direction (reversed input)
        int l = LL - 1 - s;
        gru_step(h3a, h3b, g1bWhh_bf, g1b_bhh,
                 gi2b, (long)LL * G2, (long)l * G2,
                 nullptr, nullptr, 0, 0,
                 maxbuf, 2 * HH, HH, nullptr, N2, HH);
        unsigned short* tmp = h3a; h3a = h3b; h3b = tmp;
    }

    // ---- 6) sub = relu(max), reorder (b,k)->(k,b), to bf16 -------------------
    {
        long n = (long)N2 * (2 * HH);
        maxrelu_reorder_kernel<<<(int)((n + 255) / 256), 256, 0, stream>>>(maxbuf, subT_bf);
    }

    // ---- 7) BiLSTM(384) input gates precompute -------------------------------
    gemm(subT_bf, 2 * HH, lfWih_bf, 2 * HH, lf_bih, nullptr, gi4f, nullptr, KK * BB, G4, 2 * HH);
    gemm(subT_bf, 2 * HH, lbWih_bf, 2 * HH, lb_bih, nullptr, gi4b, nullptr, KK * BB, G4, 2 * HH);

    // ---- 8) BiLSTM over K=24; feature-wise max -> ast_out --------------------
    fillf(astmax, -1e30f, (long)BB * (2 * HH));
    fillh(h4a, 0, (long)BB * HH);
    fillf(c4, 0.f, (long)BB * HH);
    for (int k = 0; k < KK; ++k) {   // forward
        lstm_step(h4a, h4b, c4, c4, lfWhh_bf, lf_bhh,
                  gi4f, (long)G4, (long)k * BB * G4,
                  astmax, 2 * HH, 0, BB, HH);
        unsigned short* tmp = h4a; h4a = h4b; h4b = tmp;
    }
    fillh(h4a, 0, (long)BB * HH);
    fillf(c4, 0.f, (long)BB * HH);
    for (int s = 0; s < KK; ++s) {   // backward (reversed input)
        int k = KK - 1 - s;
        lstm_step(h4a, h4b, c4, c4, lbWhh_bf, lb_bhh,
                  gi4b, (long)G4, (long)k * BB * G4,
                  astmax, 2 * HH, HH, BB, HH);
        unsigned short* tmp = h4a; h4a = h4b; h4b = tmp;
    }

    // ---- 9) MLP head ---------------------------------------------------------
    head_kernel<<<1, 256, 0, stream>>>(astmax, bn1_g, bn1_b, dense_W, dense_b,
                                       bn2_g, bn2_b, dense2_W, dense2_b,
                                       out_W, out_b, tmp1, tmp2, out);
}
